// Block_33981781246196
// MI455X (gfx1250) — matrix operand order repaired and checked by reading, not compiled, not run on hardware
//
#include <hip/hip_runtime.h>

typedef __bf16 bf16_t;
typedef __attribute__((ext_vector_type(16))) __bf16 v16bf;
typedef __attribute__((ext_vector_type(8)))  __bf16 v8bf;
typedef __attribute__((ext_vector_type(8)))  float  v8f;

#define BB 128
#define TT 256
#define CC 512
#define HH 8
#define HD 64
#define EE 8
#define FF 2048
#define NN (BB*TT)   // 32768 tokens

// ---------- WMMA helpers (CDNA5 wave32, 16x16x32 bf16 -> f32) ----------

__device__ __forceinline__ v8f wmma_bf16(v16bf a, v16bf b, v8f c) {
  return __builtin_amdgcn_wmma_f32_16x16x32_bf16(false, a, false, b, (short)0, c, false, false);
}

// A fragment: 16(M) x 32(K) from row-major [M x K] with leading dim ld.
// lane holds row = (lane&15); element e maps to K = (e>>3)*16 + half*8 + (e&7)
// -> per lane two contiguous 16B chunks at k0+half*8 and k0+16+half*8.
__device__ __forceinline__ v16bf frag_A(const bf16_t* base, int ld, int row, int k0, int half) {
  const bf16_t* p = base + (size_t)row * ld + k0 + half * 8;
  v8bf lo = *(const v8bf*)(p);
  v8bf hi = *(const v8bf*)(p + 16);
  v16bf r;
#pragma unroll
  for (int e = 0; e < 8; ++e) { r[e] = lo[e]; r[e + 8] = hi[e]; }
  return r;
}

// B fragment from a K-contiguous source: logical B[k][n] = src[n][k],
// src row-major [N x K] with leading dim ld (transposed weights, K^T, V^T).
// lane holds col n = (lane&15); element e maps to K = (e>>3)*16 + half*8 + (e&7)
// -> per lane two contiguous 16B chunks at k0+half*8 and k0+16+half*8.
__device__ __forceinline__ v16bf frag_Bt(const bf16_t* base, int ld, int n, int k0, int half) {
  const bf16_t* p = base + (size_t)n * ld + k0 + half * 8;
  v8bf lo = *(const v8bf*)(p);
  v8bf hi = *(const v8bf*)(p + 16);
  v16bf r;
#pragma unroll
  for (int e = 0; e < 8; ++e) { r[e] = lo[e]; r[e + 8] = hi[e]; }
  return r;
}

// ---------- small utility kernels ----------

// in: [batch][R][Cc] fp32  ->  out: [batch][Cc][R] bf16 (transpose + convert)
__global__ void transpose_cvt_kernel(const float* __restrict__ in, bf16_t* __restrict__ out,
                                     int R, int Cc, int n) {
  int o = blockIdx.x * blockDim.x + threadIdx.x;
  if (o >= n) return;
  int rc = R * Cc;
  int b = o / rc;
  int rem = o - b * rc;
  int c = rem / R;       // output row  == input col
  int r = rem - c * R;   // output col  == input row
  out[o] = (bf16_t)in[(size_t)b * rc + (size_t)r * Cc + c];
}

__global__ void zero_counts_kernel(int* counts) {
  if (threadIdx.x < EE) counts[threadIdx.x] = 0;
}

// ---------- LayerNorm 1 (fp32 stats, bf16 output) ----------

__global__ void ln1_kernel(const float* __restrict__ x, const float* __restrict__ g,
                           const float* __restrict__ b, bf16_t* __restrict__ h) {
  int tok = blockIdx.x;
  int tid = threadIdx.x;
  const float* row = x + (size_t)tok * CC;
  __shared__ float red[256];
  float v0 = row[tid], v1 = row[tid + 256];
  red[tid] = v0 + v1;
  __syncthreads();
  for (int off = 128; off > 0; off >>= 1) { if (tid < off) red[tid] += red[tid + off]; __syncthreads(); }
  float mean = red[0] * (1.0f / CC);
  __syncthreads();
  float d0 = v0 - mean, d1 = v1 - mean;
  red[tid] = d0 * d0 + d1 * d1;
  __syncthreads();
  for (int off = 128; off > 0; off >>= 1) { if (tid < off) red[tid] += red[tid + off]; __syncthreads(); }
  float rstd = rsqrtf(red[0] * (1.0f / CC) + 1e-5f);
  bf16_t* hr = h + (size_t)tok * CC;
  hr[tid]       = (bf16_t)(d0 * rstd * g[tid]       + b[tid]);
  hr[tid + 256] = (bf16_t)(d1 * rstd * g[tid + 256] + b[tid + 256]);
}

// ---------- QKV projection: per (b,h): [T x C] @ [C x HD] ----------
// Weights pre-transposed to [H][HD][C]; V written transposed [bh][HD][T].

__global__ void qkv_kernel(const bf16_t* __restrict__ h1,
                           const bf16_t* __restrict__ wqt, const bf16_t* __restrict__ wkt,
                           const bf16_t* __restrict__ wvt,
                           bf16_t* __restrict__ q, bf16_t* __restrict__ k, bf16_t* __restrict__ vt) {
  int lane = threadIdx.x & 31, half = lane >> 4, colid = lane & 15;
  int row0 = blockIdx.x * 16;
  int bh = blockIdx.y;
  int bidx = bh / HH, hidx = bh % HH;
  int which = blockIdx.z;
  const bf16_t* Wt = (which == 0 ? wqt : which == 1 ? wkt : wvt) + (size_t)hidx * HD * CC;
  const bf16_t* A = h1 + (size_t)bidx * TT * CC;

  v8f acc[4] = {};
  for (int k0 = 0; k0 < CC; k0 += 32) {
    v16bf a = frag_A(A, CC, row0 + colid, k0, half);
#pragma unroll
    for (int j = 0; j < 4; ++j) {
      v16bf bfr = frag_Bt(Wt, CC, j * 16 + colid, k0, half);
      acc[j] = wmma_bf16(a, bfr, acc[j]);
    }
  }
  if (which == 2) {   // V stored transposed: vt[bh][d][t]
    bf16_t* out = vt + (size_t)bh * HD * TT;
#pragma unroll
    for (int j = 0; j < 4; ++j)
#pragma unroll
      for (int r = 0; r < 8; ++r)
        out[(size_t)(j * 16 + colid) * TT + row0 + r + 8 * half] = (bf16_t)acc[j][r];
  } else {
    bf16_t* out = (which == 0 ? q : k) + (size_t)bh * TT * HD;
#pragma unroll
    for (int j = 0; j < 4; ++j)
#pragma unroll
      for (int r = 0; r < 8; ++r)
        out[(size_t)(row0 + r + 8 * half) * HD + j * 16 + colid] = (bf16_t)acc[j][r];
  }
}

// ---------- Flash attention (causal), one wave per 16-query tile ----------

__global__ void attn_kernel(const bf16_t* __restrict__ qb, const bf16_t* __restrict__ kb,
                            const bf16_t* __restrict__ vtb, bf16_t* __restrict__ ob) {
  int lane = threadIdx.x, half = lane >> 4, colid = lane & 15;
  int q0 = blockIdx.x * 16;
  int bh = blockIdx.y;
  const bf16_t* Q  = qb  + (size_t)bh * TT * HD;
  const bf16_t* Kp = kb  + (size_t)bh * TT * HD;
  const bf16_t* Vt = vtb + (size_t)bh * HD * TT;   // [HD][T]

  v16bf aq0 = frag_A(Q, HD, q0 + colid, 0, half);
  v16bf aq1 = frag_A(Q, HD, q0 + colid, 32, half);
  v8f oa[4] = {};
  float mrow[8], lrow[8];
#pragma unroll
  for (int r = 0; r < 8; ++r) { mrow[r] = -1e30f; lrow[r] = 0.f; }

  __shared__ __align__(16) bf16_t shp[16 * 32];

  for (int s0 = 0; s0 < q0 + 16; s0 += 32) {
    v8f sA = {}, sB = {};
    {
      v16bf b0 = frag_Bt(Kp, HD, s0 + colid, 0, half);
      v16bf b1 = frag_Bt(Kp, HD, s0 + colid, 32, half);
      sA = wmma_bf16(aq0, b0, sA);
      sA = wmma_bf16(aq1, b1, sA);
    }
    {
      v16bf b0 = frag_Bt(Kp, HD, s0 + 16 + colid, 0, half);
      v16bf b1 = frag_Bt(Kp, HD, s0 + 16 + colid, 32, half);
      sB = wmma_bf16(aq0, b0, sB);
      sB = wmma_bf16(aq1, b1, sB);
    }
#pragma unroll
    for (int r = 0; r < 8; ++r) {
      int qi = q0 + r + 8 * half;
      float s1 = sA[r] * 0.125f, s2 = sB[r] * 0.125f;
      if (s0 + colid > qi)      s1 = -1e30f;
      if (s0 + 16 + colid > qi) s2 = -1e30f;
      float mx = fmaxf(s1, s2);
#pragma unroll
      for (int msk = 8; msk >= 1; msk >>= 1) mx = fmaxf(mx, __shfl_xor(mx, msk, 32));
      float newm = fmaxf(mrow[r], mx);
      float p1 = __expf(s1 - newm), p2 = __expf(s2 - newm);
      float alpha = __expf(mrow[r] - newm);
      float ps = p1 + p2;
#pragma unroll
      for (int msk = 8; msk >= 1; msk >>= 1) ps += __shfl_xor(ps, msk, 32);
      lrow[r] = lrow[r] * alpha + ps;
      mrow[r] = newm;
#pragma unroll
      for (int j = 0; j < 4; ++j) oa[j][r] *= alpha;
      shp[(r + 8 * half) * 32 + colid]      = (bf16_t)p1;
      shp[(r + 8 * half) * 32 + colid + 16] = (bf16_t)p2;
    }
    __syncthreads();
    asm volatile("s_wait_dscnt 0" ::: "memory");   // cross-lane LDS handoff (CDNA5 split counters)
    v16bf pa = frag_A(shp, 32, colid, 0, half);
#pragma unroll
    for (int j = 0; j < 4; ++j) {
      v16bf bv = frag_Bt(Vt, TT, j * 16 + colid, s0, half);  // B[k=s][n=d] = Vt[d][s]
      oa[j] = wmma_bf16(pa, bv, oa[j]);
    }
    __syncthreads();
  }

  int bidx = bh / HH, hidx = bh % HH;
#pragma unroll
  for (int r = 0; r < 8; ++r) {
    float inv = 1.0f / lrow[r];
    int t = q0 + r + 8 * half;
    bf16_t* orow = ob + ((size_t)(bidx * TT + t)) * (HH * HD) + hidx * HD;
#pragma unroll
    for (int j = 0; j < 4; ++j)
      orow[j * 16 + colid] = (bf16_t)(oa[j][r] * inv);
  }
}

// ---------- output projection + residual: x1 = x + o @ Wo + bo ----------
// Wo pre-transposed to [C][C] (wot[n][k] = Wo[k][n]).

__global__ void oproj_kernel(const bf16_t* __restrict__ ob, const bf16_t* __restrict__ wot,
                             const float* __restrict__ bo, const float* __restrict__ x,
                             float* __restrict__ x1) {
  int lane = threadIdx.x & 31, half = lane >> 4, colid = lane & 15;
  int colbase = blockIdx.x * 64;
  int row0 = blockIdx.y * 16;
  v8f acc[4] = {};
  for (int k0 = 0; k0 < CC; k0 += 32) {
    v16bf a = frag_A(ob, CC, row0 + colid, k0, half);
#pragma unroll
    for (int j = 0; j < 4; ++j) {
      v16bf bfr = frag_Bt(wot, CC, colbase + j * 16 + colid, k0, half);
      acc[j] = wmma_bf16(a, bfr, acc[j]);
    }
  }
#pragma unroll
  for (int j = 0; j < 4; ++j) {
    int col = colbase + j * 16 + colid;
#pragma unroll
    for (int r = 0; r < 8; ++r) {
      size_t idx = (size_t)(row0 + r + 8 * half) * CC + col;
      x1[idx] = x[idx] + acc[j][r] + bo[col];
    }
  }
}

// ---------- LN2 + gate logits + top-2 routing ----------

__global__ void ln2_gate_kernel(const float* __restrict__ x1, const float* __restrict__ g,
                                const float* __restrict__ b, const float* __restrict__ gateW,
                                bf16_t* __restrict__ h2, float* __restrict__ wslot,
                                int* __restrict__ counts, int* __restrict__ lists) {
  int tok = blockIdx.x;
  int tid = threadIdx.x;
  const float* row = x1 + (size_t)tok * CC;
  __shared__ float red[256];
  __shared__ float shh[CC];
  __shared__ float shlog[EE];
  float v0 = row[tid], v1 = row[tid + 256];
  red[tid] = v0 + v1;
  __syncthreads();
  for (int off = 128; off > 0; off >>= 1) { if (tid < off) red[tid] += red[tid + off]; __syncthreads(); }
  float mean = red[0] * (1.0f / CC);
  __syncthreads();
  float d0 = v0 - mean, d1 = v1 - mean;
  red[tid] = d0 * d0 + d1 * d1;
  __syncthreads();
  for (int off = 128; off > 0; off >>= 1) { if (tid < off) red[tid] += red[tid + off]; __syncthreads(); }
  float rstd = rsqrtf(red[0] * (1.0f / CC) + 1e-5f);
  float n0 = d0 * rstd * g[tid]       + b[tid];
  float n1 = d1 * rstd * g[tid + 256] + b[tid + 256];
  shh[tid] = n0; shh[tid + 256] = n1;
  bf16_t* hr = h2 + (size_t)tok * CC;
  hr[tid] = (bf16_t)n0; hr[tid + 256] = (bf16_t)n1;
  __syncthreads();

  // 8 waves -> 8 expert logits (dot over C=512)
  int wv = tid >> 5, lane = tid & 31;
  float s = 0.f;
  for (int c = lane; c < CC; c += 32) s += shh[c] * gateW[c * EE + wv];
#pragma unroll
  for (int msk = 16; msk >= 1; msk >>= 1) s += __shfl_xor(s, msk, 32);
  if (lane == 0) shlog[wv] = s;
  __syncthreads();

  if (tid == 0) {
    float l0 = -1e30f; int i0 = 0;
#pragma unroll
    for (int e = 0; e < EE; ++e) if (shlog[e] > l0) { l0 = shlog[e]; i0 = e; }
    float l1 = -1e30f; int i1 = 0;
#pragma unroll
    for (int e = 0; e < EE; ++e) if (e != i0 && shlog[e] > l1) { l1 = shlog[e]; i1 = e; }
    // softmax over top-2 == reference's renormalized top-2 of full softmax
    float e1 = __expf(l1 - l0);
    float w0 = 1.0f / (1.0f + e1);
    float w1 = e1 / (1.0f + e1);
    int p0 = atomicAdd(&counts[i0], 1); lists[i0 * NN + p0] = tok * 2;
    int p1 = atomicAdd(&counts[i1], 1); lists[i1 * NN + p1] = tok * 2 + 1;
    wslot[tok * 2]     = w0;
    wslot[tok * 2 + 1] = w1;
  }
}

// ---------- MoE expert FFN: gathered 16-token tiles, LDS-staged hid ----------
// W1 pre-transposed to [E][FF][C], W2 pre-transposed to [E][C][FF].

__global__ void __launch_bounds__(256) moe_kernel(
    const bf16_t* __restrict__ h2, const bf16_t* __restrict__ w1t, const float* __restrict__ b1,
    const bf16_t* __restrict__ w2t, const float* __restrict__ b2, const float* __restrict__ wslot,
    const int* __restrict__ counts, const int* __restrict__ lists,
    float* __restrict__ contrib) {
  int e = blockIdx.y;
  int tile = blockIdx.x;
  int cnt = counts[e];
  if (tile * 16 >= cnt) return;

  __shared__ __align__(16) bf16_t sh_a[16 * CC];       // 16 KB gathered activations
  __shared__ __align__(16) bf16_t sh_hid[16 * 1024];   // 32 KB hid half (FF processed in 2 halves)
  __shared__ int   sh_entry[16];
  __shared__ float sh_we[16];

  int tid = threadIdx.x;
  if (tid < 16) {
    int rr = tile * 16 + tid;
    int entry = (rr < cnt) ? lists[e * NN + rr] : -1;
    sh_entry[tid] = entry;
    sh_we[tid] = (entry >= 0) ? wslot[entry] : 0.f;
  }
  __syncthreads();

  for (int i = tid; i < 16 * CC; i += 256) {
    int rr = i >> 9, c = i & (CC - 1);
    int entry = sh_entry[rr];
    sh_a[i] = (entry >= 0) ? h2[(size_t)(entry >> 1) * CC + c] : (bf16_t)0.f;
  }
  __syncthreads();

  int wv = tid >> 5, lane = tid & 31, half = lane >> 4, colid = lane & 15;
  const bf16_t* W1te = w1t + (size_t)e * FF * CC;   // [FF][C]
  const bf16_t* W2te = w2t + (size_t)e * CC * FF;   // [C][FF]

  v8f oc[4] = {};   // 64 output cols per wave (8 waves x 64 = C)
  for (int ffh = 0; ffh < 2; ++ffh) {
    // hid half: each wave owns 128 ff-cols
    v8f hacc[8] = {};
    for (int k0 = 0; k0 < CC; k0 += 32) {
      v16bf a = frag_A(sh_a, CC, colid, k0, half);
#pragma unroll
      for (int j = 0; j < 8; ++j) {
        int colg = ffh * 1024 + wv * 128 + j * 16 + colid;
        v16bf bfr = frag_Bt(W1te, CC, colg, k0, half);
        hacc[j] = wmma_bf16(a, bfr, hacc[j]);
      }
    }
#pragma unroll
    for (int j = 0; j < 8; ++j) {
      int coll = wv * 128 + j * 16 + colid;
      float bias = b1[e * FF + ffh * 1024 + coll];
#pragma unroll
      for (int r = 0; r < 8; ++r) {
        float hv = hacc[j][r] + bias;
        hv = hv > 0.f ? hv : 0.f;
        sh_hid[(r + 8 * half) * 1024 + coll] = (bf16_t)hv;
      }
    }
    __syncthreads();
    // out partial: K = this 1024-wide half
    for (int k0 = 0; k0 < 1024; k0 += 32) {
      v16bf a = frag_A(sh_hid, 1024, colid, k0, half);
#pragma unroll
      for (int j = 0; j < 4; ++j) {
        int coln = wv * 64 + j * 16 + colid;
        v16bf bfr = frag_Bt(W2te, FF, coln, ffh * 1024 + k0, half);
        oc[j] = wmma_bf16(a, bfr, oc[j]);
      }
    }
    __syncthreads();
  }

#pragma unroll
  for (int r = 0; r < 8; ++r) {
    int entry = sh_entry[r + 8 * half];
    if (entry < 0) continue;
    float we = sh_we[r + 8 * half];
    float* crow = contrib + (size_t)entry * CC;
#pragma unroll
    for (int j = 0; j < 4; ++j) {
      int coln = wv * 64 + j * 16 + colid;
      crow[coln] = we * (oc[j][r] + b2[e * CC + coln]);
    }
  }
}

// ---------- combine: out = x1 + contrib[tok,0] + contrib[tok,1] ----------

__global__ void combine_kernel(const float* __restrict__ x1, const float* __restrict__ contrib,
                               float* __restrict__ out) {
  size_t idx = (size_t)blockIdx.x * blockDim.x + threadIdx.x;
  if (idx >= (size_t)NN * CC) return;
  size_t tok = idx / CC;
  size_t col = idx - tok * CC;
  out[idx] = x1[idx] + contrib[(tok * 2) * CC + col] + contrib[(tok * 2 + 1) * CC + col];
}

// ---------- host launch ----------

extern "C" void kernel_launch(void* const* d_in, const int* in_sizes, int n_in,
                              void* d_out, int out_size, void* d_ws, size_t ws_size,
                              hipStream_t stream) {
  (void)in_sizes; (void)n_in; (void)out_size; (void)ws_size;
  const float* x     = (const float*)d_in[0];
  const float* ln1_g = (const float*)d_in[1];
  const float* ln1_b = (const float*)d_in[2];
  const float* Wq    = (const float*)d_in[3];
  const float* Wk    = (const float*)d_in[4];
  const float* Wv    = (const float*)d_in[5];
  const float* Wo    = (const float*)d_in[6];
  const float* bo    = (const float*)d_in[7];
  const float* ln2_g = (const float*)d_in[8];
  const float* ln2_b = (const float*)d_in[9];
  const float* gateW = (const float*)d_in[10];
  const float* W1    = (const float*)d_in[11];
  const float* b1    = (const float*)d_in[12];
  const float* W2    = (const float*)d_in[13];
  const float* b2    = (const float*)d_in[14];
  float* out = (float*)d_out;

  char* base = (char*)d_ws;
  size_t off = 0;
  auto alloc = [&](size_t bytes) -> char* {
    char* p = base + off;
    off = (off + bytes + 255) & ~(size_t)255;
    return p;
  };

  bf16_t* h1   = (bf16_t*)alloc((size_t)NN * CC * 2);    // reused as h2 after attention
  char*   qkvo = alloc((size_t)4 * NN * CC * 2);          // q,k,vt,o ; reused as MoE slot buffer
  bf16_t* qb   = (bf16_t*)qkvo;
  bf16_t* kb   = qb + (size_t)NN * CC;
  bf16_t* vtb  = kb + (size_t)NN * CC;
  bf16_t* obuf = vtb + (size_t)NN * CC;
  float*  contrib = (float*)qkvo;                         // NN*2 slots x C fp32 == same 128 MB
  float*  x1   = (float*)alloc((size_t)NN * CC * 4);
  bf16_t* wqt  = (bf16_t*)alloc((size_t)HH * CC * HD * 2);
  bf16_t* wkt  = (bf16_t*)alloc((size_t)HH * CC * HD * 2);
  bf16_t* wvt  = (bf16_t*)alloc((size_t)HH * CC * HD * 2);
  bf16_t* wot  = (bf16_t*)alloc((size_t)CC * CC * 2);
  bf16_t* w1t  = (bf16_t*)alloc((size_t)EE * CC * FF * 2);
  bf16_t* w2t  = (bf16_t*)alloc((size_t)EE * FF * CC * 2);
  float*  wslot  = (float*)alloc((size_t)NN * 2 * 4);
  int*    counts = (int*)alloc(EE * 4);
  int*    lists  = (int*)alloc((size_t)EE * NN * 4);

  // weight transpose+convert: W^T stored [N][K] so B-fragments load K-contiguous
  int nqkvw = HH * CC * HD;
  transpose_cvt_kernel<<<(nqkvw + 255) / 256, 256, 0, stream>>>(Wq, wqt, CC, HD, nqkvw);
  transpose_cvt_kernel<<<(nqkvw + 255) / 256, 256, 0, stream>>>(Wk, wkt, CC, HD, nqkvw);
  transpose_cvt_kernel<<<(nqkvw + 255) / 256, 256, 0, stream>>>(Wv, wvt, CC, HD, nqkvw);
  int nwo = CC * CC;
  transpose_cvt_kernel<<<(nwo + 255) / 256, 256, 0, stream>>>(Wo, wot, CC, CC, nwo);
  int nmw = EE * CC * FF;
  transpose_cvt_kernel<<<(nmw + 255) / 256, 256, 0, stream>>>(W1, w1t, CC, FF, nmw);
  transpose_cvt_kernel<<<(nmw + 255) / 256, 256, 0, stream>>>(W2, w2t, FF, CC, nmw);

  zero_counts_kernel<<<1, 32, 0, stream>>>(counts);

  ln1_kernel<<<NN, 256, 0, stream>>>(x, ln1_g, ln1_b, h1);
  qkv_kernel<<<dim3(TT / 16, BB * HH, 3), 32, 0, stream>>>(h1, wqt, wkt, wvt, qb, kb, vtb);
  attn_kernel<<<dim3(TT / 16, BB * HH), 32, 0, stream>>>(qb, kb, vtb, obuf);
  oproj_kernel<<<dim3(CC / 64, NN / 16), 32, 0, stream>>>(obuf, wot, bo, x, x1);
  ln2_gate_kernel<<<NN, 256, 0, stream>>>(x1, ln2_g, ln2_b, gateW, h1, wslot, counts, lists);
  moe_kernel<<<dim3(NN / 16, EE), 256, 0, stream>>>(h1, w1t, b1, w2t, b2, wslot, counts, lists, contrib);
  combine_kernel<<<(int)(((size_t)NN * CC + 255) / 256), 256, 0, stream>>>(x1, contrib, out);
}
